// LargeKernelAttn_4939212390843
// MI455X (gfx1250) — matrix operand order repaired and checked by reading, not compiled, not run on hardware
//
#include <hip/hip_runtime.h>

typedef __attribute__((ext_vector_type(16))) _Float16 v16h;
typedef __attribute__((ext_vector_type(8)))  _Float16 v8h;
typedef __attribute__((ext_vector_type(8)))  float    v8f;

#define C64   64
#define IMG   256
#define HW    65536      // 256*256
#define TILE  64

// ---------------------------------------------------------------------------
// Kernel A: fused depthwise 5x5 (pad2) -> depthwise 7x7 dil3 (pad9), per
// (batch, channel, 64x64 tile). Output h2 stored f16, pixel-major [p][c].
// Inner loops restructured into multiple independent FMA chains (VOPD
// dual-issue) with shared LDS taps.
// ---------------------------------------------------------------------------
__global__ __launch_bounds__(256) void LKA_dw_fused(
    const float* __restrict__ x,
    const float* __restrict__ w5g, const float* __restrict__ b5g,
    const float* __restrict__ w7g, const float* __restrict__ b7g,
    _Float16* __restrict__ h2t)
{
    __shared__ float sIn[86 * 86];   // input tile + halo 11
    __shared__ float sH1[82 * 82];   // h1 tile + halo 9
    __shared__ float sW5[25];
    __shared__ float sW7[49];
    __shared__ float sBias[2];

    const int tid = threadIdx.x;
    const int blk = blockIdx.x;
    const int tx = (blk & 3) * TILE;
    const int ty = ((blk >> 2) & 3) * TILE;
    const int c  = (blk >> 4) & 63;
    const int b  = blk >> 10;

    if (tid < 25)       sW5[tid]        = w5g[c * 25 + tid];
    else if (tid < 74)  sW7[tid - 25]   = w7g[c * 49 + (tid - 25)];
    else if (tid == 74) sBias[0]        = b5g[c];
    else if (tid == 75) sBias[1]        = b7g[c];

    const float* xc = x + (b * C64 + c) * HW;

    // Stage input tile (zero-padded outside the image: conv1 pad=2 region).
    for (int i = tid; i < 86 * 86; i += 256) {
        int gy = ty + (i / 86) - 11;
        int gx = tx + (i % 86) - 11;
        float v = 0.0f;
        if ((unsigned)gy < (unsigned)IMG && (unsigned)gx < (unsigned)IMG)
            v = xc[gy * IMG + gx];
        sIn[i] = v;
    }
    __syncthreads();

    // h1 = dw5x5(x) + b1 on 82x82 halo region, 2 adjacent columns per thread
    // (two independent FMA chains, taps shared). Out-of-image h1 forced to 0
    // because conv2 zero-pads h1 itself (bias must NOT leak outside).
    const float bias1 = sBias[0];
    for (int i = tid; i < 82 * 41; i += 256) {
        int p = i / 41, q = (i % 41) * 2;
        float a0 = bias1, a1 = bias1;
#pragma unroll
        for (int di = 0; di < 5; ++di) {
            const float* r = &sIn[(p + di) * 86 + q];
            float t0 = r[0], t1 = r[1], t2 = r[2], t3 = r[3], t4 = r[4], t5 = r[5];
            const float* w = &sW5[di * 5];
            a0 = fmaf(w[0], t0, a0);  a1 = fmaf(w[0], t1, a1);
            a0 = fmaf(w[1], t1, a0);  a1 = fmaf(w[1], t2, a1);
            a0 = fmaf(w[2], t2, a0);  a1 = fmaf(w[2], t3, a1);
            a0 = fmaf(w[3], t3, a0);  a1 = fmaf(w[3], t4, a1);
            a0 = fmaf(w[4], t4, a0);  a1 = fmaf(w[4], t5, a1);
        }
        int ay = ty + p - 9;
        int ax = tx + q - 9;
        bool vy = (unsigned)ay < (unsigned)IMG;
        sH1[p * 82 + q]     = (vy && (unsigned)ax       < (unsigned)IMG) ? a0 : 0.0f;
        sH1[p * 82 + q + 1] = (vy && (unsigned)(ax + 1) < (unsigned)IMG) ? a1 : 0.0f;
    }
    __syncthreads();

    // h2 = dw7x7_dil3(h1) + b2 ; 4 adjacent columns per thread: the 7 dilated
    // taps of 4 outputs cover one contiguous 22-float run per row (shared,
    // b64-vectorizable LDS loads; 4 independent FMA chains).
    const float bias2 = sBias[1];
    for (int i = tid; i < 64 * 16; i += 256) {
        int yy = i >> 4, xg = (i & 15) * 4;
        float a0 = bias2, a1 = bias2, a2 = bias2, a3 = bias2;
#pragma unroll
        for (int di = 0; di < 7; ++di) {
            const float* r = &sH1[(yy + 3 * di) * 82 + xg];
            float t[22];
#pragma unroll
            for (int j = 0; j < 22; ++j) t[j] = r[j];
            const float* w = &sW7[di * 7];
#pragma unroll
            for (int dj = 0; dj < 7; ++dj) {
                float wv = w[dj];
                a0 = fmaf(wv, t[3 * dj + 0], a0);
                a1 = fmaf(wv, t[3 * dj + 1], a1);
                a2 = fmaf(wv, t[3 * dj + 2], a2);
                a3 = fmaf(wv, t[3 * dj + 3], a3);
            }
        }
        int p = (b << 16) + (ty + yy) * IMG + (tx + xg);   // flat pixel id
        h2t[(p + 0) * C64 + c] = (_Float16)a0;
        h2t[(p + 1) * C64 + c] = (_Float16)a1;
        h2t[(p + 2) * C64 + c] = (_Float16)a2;
        h2t[(p + 3) * C64 + c] = (_Float16)a3;
    }
}

// ---------------------------------------------------------------------------
// Kernel B: pointwise 64x64 channel mix via v_wmma_f32_16x16x32_f16 + gate.
// Each wave: 16 pixels x 64 out-channels = 4 m-tiles x 2 k-chunks = 8 WMMAs.
// D layout: M = out-channel (tile), N = pixel  -> coalesced stores on lanes.
// ---------------------------------------------------------------------------
__global__ __launch_bounds__(256) void LKA_pw_wmma(
    const float* __restrict__ x,
    const _Float16* __restrict__ h2t,
    const float* __restrict__ pw_w,
    const float* __restrict__ pw_b,
    float* __restrict__ out)
{
    __shared__ _Float16 sW[C64 * C64];   // W[o][c] as f16, 8 KB
    __shared__ float    sB[C64];

    const int tid = threadIdx.x;
    for (int i = tid; i < C64 * C64; i += 256) sW[i] = (_Float16)pw_w[i];
    if (tid < C64) sB[tid] = pw_b[tid];
    __syncthreads();

    const int wave = tid >> 5;
    const int lane = tid & 31;
    const int lp   = lane & 15;   // pixel slot within 16-wide tile (N)
    const int lh   = lane >> 4;   // lane half selector

    const int pixBase = blockIdx.x * 128 + wave * 16;   // 8 waves * 16 px
    const int b  = pixBase >> 16;                       // image (HW = 65536)
    const int hw = pixBase & (HW - 1);

    // B fragments: h2t[pixel][channel] f16; lane holds K = lh*16 + t, contiguous.
    const _Float16* hp = h2t + (pixBase + lp) * C64 + lh * 16;
    v16h bfrag0 = *(const v16h*)(hp);        // channels  0..31 chunk
    v16h bfrag1 = *(const v16h*)(hp + 32);   // channels 32..63 chunk

    // Hoist all 8 A-fragments so the WMMA chain can issue without per-step
    // ds waits. Lanes 0-15 hold K rel {0..15}; lanes 16-31 {16..31}.
    v16h afrag[8];
#pragma unroll
    for (int m = 0; m < 4; ++m) {
#pragma unroll
        for (int k = 0; k < 2; ++k) {
            const _Float16* wrow = sW + (m * 16 + lp) * C64 + k * 32 + lh * 16;
            v8h lo = *(const v8h*)(wrow);
            v8h hi = *(const v8h*)(wrow + 8);
            afrag[m * 2 + k] = __builtin_shufflevector(lo, hi,
                      0, 1, 2, 3, 4, 5, 6, 7, 8, 9, 10, 11, 12, 13, 14, 15);
        }
    }

    v8f acc[4] = {v8f{}, v8f{}, v8f{}, v8f{}};
#pragma unroll
    for (int m = 0; m < 4; ++m) {
        acc[m] = __builtin_amdgcn_wmma_f32_16x16x32_f16(
                     false, afrag[m * 2 + 0], false, bfrag0,
                     (short)0, acc[m], false, false);
        acc[m] = __builtin_amdgcn_wmma_f32_16x16x32_f16(
                     false, afrag[m * 2 + 1], false, bfrag1,
                     (short)0, acc[m], false, false);
    }

    // weight = acc + bias ; out = x * weight. M = lh*8 + r within tile.
    const int base = (b * C64) * HW + hw + lp;
#pragma unroll
    for (int m = 0; m < 4; ++m) {
#pragma unroll
        for (int r = 0; r < 8; ++r) {
            const int o   = m * 16 + lh * 8 + r;
            const int idx = base + o * HW;
            out[idx] = x[idx] * (acc[m][r] + sB[o]);
        }
    }
}

// ---------------------------------------------------------------------------
extern "C" void kernel_launch(void* const* d_in, const int* in_sizes, int n_in,
                              void* d_out, int out_size, void* d_ws, size_t ws_size,
                              hipStream_t stream)
{
    const float* x     = (const float*)d_in[0];
    const float* dw_w  = (const float*)d_in[1];
    const float* dw_b  = (const float*)d_in[2];
    const float* dwd_w = (const float*)d_in[3];
    const float* dwd_b = (const float*)d_in[4];
    const float* pw_w  = (const float*)d_in[5];
    const float* pw_b  = (const float*)d_in[6];
    float* out = (float*)d_out;

    // Scratch: h2 intermediate, f16 pixel-major [B*H*W][64] = 134 MB.
    _Float16* h2t = (_Float16*)d_ws;

    // 16 batches * 64 channels * (4x4) 64x64 tiles = 16384 blocks.
    LKA_dw_fused<<<16384, 256, 0, stream>>>(x, dw_w, dw_b, dwd_w, dwd_b, h2t);

    // 1,048,576 pixels / (8 waves * 16 px) = 8192 blocks.
    LKA_pw_wmma<<<8192, 256, 0, stream>>>(x, h2t, pw_w, pw_b, out);
}
